// LocalAggregator_63642825392518
// MI455X (gfx1250) — matrix operand order repaired and checked by reading, not compiled, not run on hardware
//
#include <hip/hip_runtime.h>
#include <hip/hip_bf16.h>

typedef _Float16 v16h __attribute__((ext_vector_type(16)));
typedef _Float16 v8h  __attribute__((ext_vector_type(8)));
typedef float    v8f  __attribute__((ext_vector_type(8)));

#define BB   32
#define NN   256
#define DD   128
#define HOPS 2
#define LRA  0.2f

__device__ __forceinline__ v16h cat8(v8h lo, v8h hi) {
  return __builtin_shufflevector(lo, hi, 0,1,2,3,4,5,6,7,8,9,10,11,12,13,14,15);
}

__global__ __launch_bounds__(128)
void LocalAggregator_63642825392518_kernel(const float* __restrict__ hidden,
                                           const int*   __restrict__ adj,
                                           const float* __restrict__ a,
                                           float*       __restrict__ out) {
  extern __shared__ char smem[];
  _Float16* Hs  = (_Float16*)smem;            // 256x128 row-major f16 (64 KB)
  _Float16* HTs = Hs  + NN * DD;              // 128x256 transposed   (64 KB)
  _Float16* Ts  = HTs + NN * DD;              // 4 waves x 16x256 tmp (32 KB)
  _Float16* As  = Ts  + 4 * 16 * NN;          // 2x128 hop vectors    (512 B)

  const int tid  = threadIdx.x;
  const int lane = tid & 31;
  const int wv   = tid >> 5;
  const int b    = blockIdx.x >> 2;
  const int i0   = (blockIdx.x & 3) * 64 + wv * 16;

  // ---- stage hidden[b] into LDS (f16), both row-major and transposed ----
  const float* hb = hidden + b * (NN * DD);
  for (int e = tid * 4; e < NN * DD; e += 128 * 4) {
    float4 v = *(const float4*)(hb + e);
    int row = e >> 7, col = e & (DD - 1);
    float vv[4] = {v.x, v.y, v.z, v.w};
#pragma unroll
    for (int k = 0; k < 4; ++k) {
      _Float16 hv = (_Float16)vv[k];
      Hs [row * DD + col + k]   = hv;
      HTs[(col + k) * NN + row] = hv;
    }
  }
  for (int t = tid; t < HOPS * DD; t += 128) As[t] = (_Float16)a[t];
  __syncthreads();

  const int half  = lane >> 4;     // 0 or 1 (which 16-lane half)
  const int l15   = lane & 15;
  const int arow  = i0 + l15;      // GEMM1 A-fragment row
  const int rbase = half * 8;      // C-layout row offset for this half

  // ---- hoist GEMM1 A-fragments: (a_h . H)[i-tile, :] for both hops, all K ----
  // Invariant across all 16 j-tiles; 8 x v16h = 64 VGPRs, freed before phase 2.
  v16h A0k[4], A1k[4];
#pragma unroll
  for (int kk = 0; kk < 4; ++kk) {
    const int ka = kk * 32 + (half << 3);
    v8h h1  = *(const v8h*)&Hs[arow * DD + ka];
    v8h h2  = *(const v8h*)&Hs[arow * DD + ka + 16];
    v8h a0l = *(const v8h*)&As[ka];
    v8h a0h = *(const v8h*)&As[ka + 16];
    v8h a1l = *(const v8h*)&As[DD + ka];
    v8h a1h = *(const v8h*)&As[DD + ka + 16];
    A0k[kk] = cat8(h1 * a0l, h2 * a0h);
    A1k[kk] = cat8(h1 * a1l, h2 * a1h);
  }

  _Float16* Tw = Ts + wv * (16 * NN);
  float sPart[8];
#pragma unroll
  for (int r = 0; r < 8; ++r) sPart[r] = 0.f;

  // ---- phase 1: E_h = (a_h . H) H^T ; lrelu, mask by adj, exp ; tmp -> LDS ----
  for (int jt = 0; jt < 16; ++jt) {
    const int j0 = jt * 16;
    const int jj = j0 + l15;

    // issue adj loads early so they overlap the WMMA k-loop
    const int base0 = ((b * HOPS + 0) * NN + i0 + rbase) * NN + jj;
    const int base1 = ((b * HOPS + 1) * NN + i0 + rbase) * NN + jj;
    int m0[8], m1[8];
#pragma unroll
    for (int r = 0; r < 8; ++r) {
      m0[r] = adj[base0 + r * NN];
      m1[r] = adj[base1 + r * NN];
    }

    v8f acc0 = {}; v8f acc1 = {};
#pragma unroll
    for (int kk = 0; kk < 4; ++kk) {
      // B fragment: lane = column j, K = d  (row of Hs, contiguous)
      const int kb = kk * 32 + (half << 3);
      v16h Bf = cat8(*(const v8h*)&Hs[(j0 + l15) * DD + kb],
                     *(const v8h*)&Hs[(j0 + l15) * DD + kb + 16]);
      acc0 = __builtin_amdgcn_wmma_f32_16x16x32_f16(false, A0k[kk], false, Bf, (short)0, acc0, false, false);
      acc1 = __builtin_amdgcn_wmma_f32_16x16x32_f16(false, A1k[kk], false, Bf, (short)0, acc1, false, false);
    }

    // lrelu + adj mask + exp, accumulate row partial sums, stash tmp tile as f16
#pragma unroll
    for (int r = 0; r < 8; ++r) {
      float e0 = acc0[r]; e0 = e0 > 0.f ? e0 : LRA * e0;
      float e1 = acc1[r]; e1 = e1 > 0.f ? e1 : LRA * e1;
      float t0 = (m0[r] == 1) ? __expf(e0) : 0.f;
      float t1 = (m1[r] == 2) ? __expf(e1) : 0.f;
      float t  = t0 + t1;
      sPart[r] += t;
      Tw[(rbase + r) * NN + jj] = (_Float16)t;
    }
  }
  __syncthreads();

  // ---- row sums: butterfly over the 16 lanes of each half (matches C-layout) ----
  float inv[8];
#pragma unroll
  for (int r = 0; r < 8; ++r) {
    float s = sPart[r];
    s += __shfl_xor(s, 1, 16);
    s += __shfl_xor(s, 2, 16);
    s += __shfl_xor(s, 4, 16);
    s += __shfl_xor(s, 8, 16);
    inv[r] = (s == 0.f) ? 1.f : (1.f / s);
  }

  // ---- phase 2: out = (tmp @ H) * inv ; A from tmp strip, B from H^T ----
  v8f o[8] = {};
  for (int j0 = 0; j0 < NN; j0 += 32) {
    const int ka = j0 + (half << 3);
    v16h Af = cat8(*(const v8h*)&Tw[l15 * NN + ka],
                   *(const v8h*)&Tw[l15 * NN + ka + 16]);
    const int kb = j0 + (half << 3);
#pragma unroll
    for (int dt = 0; dt < 8; ++dt) {
      const int drow = dt * 16 + l15;
      v16h Bf = cat8(*(const v8h*)&HTs[drow * NN + kb],
                     *(const v8h*)&HTs[drow * NN + kb + 16]);
      o[dt] = __builtin_amdgcn_wmma_f32_16x16x32_f16(false, Af, false, Bf, (short)0, o[dt], false, false);
    }
  }

  // ---- normalize and store (f32, coalesced per 16-lane half) ----
  float* ob = out + (size_t)(b * NN + i0 + rbase) * DD;
#pragma unroll
  for (int dt = 0; dt < 8; ++dt) {
    const int col = dt * 16 + l15;
#pragma unroll
    for (int r = 0; r < 8; ++r) {
      ob[r * DD + col] = o[dt][r] * inv[r];
    }
  }
}

extern "C" void kernel_launch(void* const* d_in, const int* in_sizes, int n_in,
                              void* d_out, int out_size, void* d_ws, size_t ws_size,
                              hipStream_t stream) {
  (void)in_sizes; (void)n_in; (void)out_size; (void)d_ws; (void)ws_size;
  const float* hidden = (const float*)d_in[0];
  const int*   adj    = (const int*)d_in[1];   // jax default x64-off: adj is int32
  const float* a      = (const float*)d_in[2];
  float* out = (float*)d_out;

  const size_t lds = (size_t)(NN * DD            /* Hs  */
                            + NN * DD            /* HTs */
                            + 4 * 16 * NN        /* Ts  */
                            + HOPS * DD) * 2;    /* As  */  // ~160.5 KB
  dim3 grid(BB * 4);
  dim3 block(128);
  LocalAggregator_63642825392518_kernel<<<grid, block, lds, stream>>>(hidden, adj, a, out);
}
